// FaceQAConv_46488726012177
// MI455X (gfx1250) — matrix operand order repaired and checked by reading, not compiled, not run on hardware
//
#include <hip/hip_runtime.h>

typedef __attribute__((ext_vector_type(16))) __bf16 v16bf;
typedef __attribute__((ext_vector_type(8)))  float  v8f;

#define NP 48
#define NG 48
#define NC 64
#define HW 256
#define NPAIR (NP*NG)
#define EPSF 1e-5f

// ds_swizzle_b32 group-of-32 XOR patterns: imm = (xor_mask<<10) | and_mask(0x1f)
#define SWZ_XOR1  0x041f
#define SWZ_XOR2  0x081f
#define SWZ_XOR4  0x101f
#define SWZ_XOR8  0x201f
#define SWZ_XOR16 0x401f
#define SWZ_MAX(v, imm) \
    fmaxf((v), __int_as_float(__builtin_amdgcn_ds_swizzle(__float_as_int(v), (imm))))

// Workspace layout (bytes). One fragment = 32 lanes * 32 B = 1024 B.
// A frags (from gal_fea): 48 obj * 16 tiles * 2 kchunks = 1536 frags.
// B frags (from prob_fea): same count.
#define SZ_HALF (1536u * 1024u)          // 1.5 MB per (matrix, hi/lo) plane
#define OFF_A_HI 0u
#define OFF_A_LO (SZ_HALF)
#define OFF_B_HI (2u * SZ_HALF)
#define OFF_B_LO (3u * SZ_HALF)
#define OFF_PART (4u * SZ_HALF)          // 2304 * 3 floats

// ---------------------------------------------------------------------------
// Kernel 1: split fp32 -> (bf16 hi, bf16 lo) and pre-swizzle into the CDNA5
// WMMA 16x16x32 bf16 register layouts so the GEMM kernel does pure B128 loads.
// A'[m][k] = gal[g][k][r0+m]   (A 16x32, m=lane&15, k=(e&7)+16*(e>>3)+8*half)
// B'[k][n] = prob[p][k][s0+n]  (B 32x16, n=lane&15, k=(e&7)+16*(e>>3)+8*half)
// ---------------------------------------------------------------------------
__global__ __launch_bounds__(256) void repack_kernel(const float* __restrict__ prob,
                                                     const float* __restrict__ gal,
                                                     char* __restrict__ ws) {
    int tid  = blockIdx.x * 256 + threadIdx.x;   // 98304 threads total
    int lane = tid & 31;
    int fi   = tid >> 5;                          // 0..3071
    bool isB = (fi >= 1536);
    int  f   = isB ? (fi - 1536) : fi;            // frag index within plane
    int  kc  = f & 1;                             // k-chunk (0: k 0..31, 1: k 32..63)
    int  T   = (f >> 1) & 15;                     // tile index along r (A) or s (B)
    int  obj = f >> 5;                            // g (A) or p (B)
    const float* src = isB ? prob : gal;

    int half = lane >> 4;
    int mn   = lane & 15;
    int c0   = 32 * kc;
    int rs0  = 16 * T + mn;

    v16bf vh, vl;
#pragma unroll
    for (int e = 0; e < 16; ++e) {
        int k = (e & 7) + 16 * (e >> 3) + 8 * half;
        float x = src[((size_t)obj * NC + (c0 + k)) * HW + rs0];
        __bf16 h = (__bf16)x;
        vh[e] = h;
        vl[e] = (__bf16)(x - (float)h);
    }
    size_t base = (size_t)f * 1024u + (size_t)lane * 32u;
    *(v16bf*)(ws + (isB ? OFF_B_HI : OFF_A_HI) + base) = vh;
    *(v16bf*)(ws + (isB ? OFF_B_LO : OFF_A_LO) + base) = vl;
}

// ---------------------------------------------------------------------------
// Kernel 2: one block per (p,g). 8 waves, each wave owns 2 tile-rows of the
// 16x16 grid of 16x16 WMMA tiles.
//
// Band pruning: the PART=4 window pools only read score tiles with
// TR in win(TC) (feeds s1 column-max) or TC in win(TR) (feeds s2 row-max);
// 78 of 256 tiles. Skip predicate depends only on (TR,TC): wave-uniform.
//
// Per needed tile: 6x v_wmma_f32_16x16x32_bf16 (bf16x3 fp32 emulation, K=64).
// Cross-lane max reductions use single-instruction ds_swizzle XOR patterns.
// ---------------------------------------------------------------------------
__global__ __launch_bounds__(256) void score_kernel(const float* __restrict__ fc_w,
                                                    const char*  __restrict__ ws,
                                                    float* __restrict__ part) {
    __shared__ float smem[8192];                // colmax[0..4095], rowmax[4096..8191]
    int b    = blockIdx.x;                      // pair index = pi*48 + gi
    int pi   = b / NG;
    int gi   = b % NG;
    int t    = threadIdx.x;
    int wv   = t >> 5;
    int lane = t & 31;

    for (int tri = 0; tri < 2; ++tri) {
        int TR = wv + 8 * tri;
        int uR = TR - 2; if (uR < 0) uR = 0; if (uR > 12) uR = 12;  // win(TR) start
        size_t fa   = ((size_t)gi * 16 + TR) * 2;
        size_t alo  = fa * 1024u + (size_t)lane * 32u;
        v16bf a0h = *(const v16bf*)(ws + OFF_A_HI + alo);
        v16bf a1h = *(const v16bf*)(ws + OFF_A_HI + alo + 1024u);
        v16bf a0l = *(const v16bf*)(ws + OFF_A_LO + alo);
        v16bf a1l = *(const v16bf*)(ws + OFF_A_LO + alo + 1024u);

        for (int TC = 0; TC < 16; ++TC) {
            int uC = TC - 2; if (uC < 0) uC = 0; if (uC > 12) uC = 12;  // win(TC) start
            bool colNeeded = (TR >= uC) && (TR < uC + 4);   // tile feeds s1[TC,:]
            bool rowNeeded = (TC >= uR) && (TC < uR + 4);   // tile feeds s2[TR,:]
            if (!(colNeeded || rowNeeded)) continue;        // dead tile: skip GEMM

            size_t fb  = ((size_t)pi * 16 + TC) * 2;
            size_t blo = fb * 1024u + (size_t)lane * 32u;
            v16bf b0h = *(const v16bf*)(ws + OFF_B_HI + blo);
            v16bf b1h = *(const v16bf*)(ws + OFF_B_HI + blo + 1024u);
            v16bf b0l = *(const v16bf*)(ws + OFF_B_LO + blo);
            v16bf b1l = *(const v16bf*)(ws + OFF_B_LO + blo + 1024u);

            v8f acc = {};
            acc = __builtin_amdgcn_wmma_f32_16x16x32_bf16(false, a0h, false, b0h, (short)0, acc, false, false);
            acc = __builtin_amdgcn_wmma_f32_16x16x32_bf16(false, a1h, false, b1h, (short)0, acc, false, false);
            acc = __builtin_amdgcn_wmma_f32_16x16x32_bf16(false, a0h, false, b0l, (short)0, acc, false, false);
            acc = __builtin_amdgcn_wmma_f32_16x16x32_bf16(false, a1h, false, b1l, (short)0, acc, false, false);
            acc = __builtin_amdgcn_wmma_f32_16x16x32_bf16(false, a0l, false, b0h, (short)0, acc, false, false);
            acc = __builtin_amdgcn_wmma_f32_16x16x32_bf16(false, a1l, false, b1h, (short)0, acc, false, false);

            if (colNeeded) {
                // Column max of the tile: per lane, max of the 8 accumulator rows
                // (lane<16: M 0..7; lane>=16: M 8..15), then merge lane ^ 16.
                float cm = acc[0];
#pragma unroll
                for (int j = 1; j < 8; ++j) cm = fmaxf(cm, acc[j]);
                cm = SWZ_MAX(cm, SWZ_XOR16);
                if (lane < 16) smem[TR * 256 + TC * 16 + lane] = cm;
            }

            if (rowNeeded) {
                // Row max of the tile: ds_swizzle butterfly over the 16-lane
                // group per accumulator register (full EXEC here).
                float rm[8];
#pragma unroll
                for (int j = 0; j < 8; ++j) {
                    float v = acc[j];
                    v = SWZ_MAX(v, SWZ_XOR1);
                    v = SWZ_MAX(v, SWZ_XOR2);
                    v = SWZ_MAX(v, SWZ_XOR4);
                    v = SWZ_MAX(v, SWZ_XOR8);
                    rm[j] = v;
                }
                if ((lane & 15) == 0) {
                    int rbase = (lane >> 4) * 8;  // lane0 -> rows 0..7, lane16 -> rows 8..15
#pragma unroll
                    for (int j = 0; j < 8; ++j)
                        smem[4096 + TR * 256 + TC * 16 + rbase + j] = rm[j];
                }
            }
        }
    }
    __syncthreads();

    // Window pooling: windows are 4 consecutive blocks, u = clamp(i-2, 0, 12).
    // Only entries with TR in win(TC) / TC in win(TR) are read -> all written.
    int blkI = t >> 4, off = t & 15;
    int u = blkI - 2; if (u < 0) u = 0; if (u > 12) u = 12;
    float s1v = -__builtin_inff(), s2v = -__builtin_inff();
#pragma unroll
    for (int i = 0; i < 4; ++i) {
        s1v = fmaxf(s1v, smem[(u + i) * 256 + blkI * 16 + off]);        // max over hr window
        s2v = fmaxf(s2v, smem[4096 + blkI * 256 + (u + i) * 16 + off]); // max over hc window
    }
    float wt = fc_w[t];
    __syncthreads();

    // Block-reduce the 3 scalars the (linear) BN+fc epilogue needs.
    smem[t]       = (s1v + s2v) * wt;       // weighted sum (fc numerator)
    smem[256 + t] = s1v + s2v;              // Σx   (for global BN mean)
    smem[512 + t] = s1v * s1v + s2v * s2v;  // Σx²  (for global BN var)
    __syncthreads();
    for (int s = 128; s > 0; s >>= 1) {
        if (t < s) {
            smem[t]       += smem[t + s];
            smem[256 + t] += smem[256 + t + s];
            smem[512 + t] += smem[512 + t + s];
        }
        __syncthreads();
    }
    if (t == 0) {
        part[b * 3 + 0] = smem[0];
        part[b * 3 + 1] = smem[256];
        part[b * 3 + 2] = smem[512];
    }
}

// ---------------------------------------------------------------------------
// Kernel 3: single block. Global BN stats over all 2*2304*256 pooled values,
// per-pair fc+sum, second BN over the 2304 scores, sigmoid. Deterministic.
// ---------------------------------------------------------------------------
__global__ __launch_bounds__(256) void finish_kernel(const float* __restrict__ part,
                                                     const float* __restrict__ fc_w,
                                                     const float* __restrict__ bn_g,
                                                     const float* __restrict__ bn_b,
                                                     const float* __restrict__ fc_b,
                                                     const float* __restrict__ lbn_g,
                                                     const float* __restrict__ lbn_b,
                                                     float* __restrict__ out) {
    __shared__ float red[768];
    __shared__ float cst[2];
    int t = threadIdx.x;

    float gs = 0.f, gss = 0.f;
    for (int i = t; i < NPAIR; i += 256) { gs += part[i * 3 + 1]; gss += part[i * 3 + 2]; }
    red[t] = gs; red[256 + t] = gss; red[512 + t] = fc_w[t];
    __syncthreads();
    for (int s = 128; s > 0; s >>= 1) {
        if (t < s) {
            red[t] += red[t + s];
            red[256 + t] += red[256 + t + s];
            red[512 + t] += red[512 + t + s];
        }
        __syncthreads();
    }
    if (t == 0) {
        float N  = 2.0f * (float)NPAIR * 256.0f;
        float M  = red[0] / N;
        float V  = red[256] / N - M * M;
        float rs = rsqrtf(V + EPSF);
        float Ws = red[512];
        float K1 = bn_g[0] * rs;
        float K0 = 2.0f * (-K1 * M * Ws + bn_b[0] * Ws + fc_b[0]);
        cst[0] = K1; cst[1] = K0;
    }
    __syncthreads();
    float K1 = cst[0], K0 = cst[1];

    float ts = 0.f, tss = 0.f;
    for (int i = t; i < NPAIR; i += 256) {
        float x = K1 * part[i * 3 + 0] + K0;
        ts += x; tss += x * x;
    }
    __syncthreads();
    red[t] = ts; red[256 + t] = tss;
    __syncthreads();
    for (int s = 128; s > 0; s >>= 1) {
        if (t < s) { red[t] += red[t + s]; red[256 + t] += red[256 + t + s]; }
        __syncthreads();
    }
    if (t == 0) {
        float mt = red[0] / (float)NPAIR;
        float vt = red[256] / (float)NPAIR - mt * mt;
        cst[0] = mt; cst[1] = rsqrtf(vt + EPSF);
    }
    __syncthreads();
    float mt = cst[0], rvt = cst[1];
    float lg = lbn_g[0], lb = lbn_b[0];
    for (int i = t; i < NPAIR; i += 256) {
        float x = K1 * part[i * 3 + 0] + K0;
        float y = lg * (x - mt) * rvt + lb;
        out[i] = 1.0f / (1.0f + __expf(-y));
    }
}

extern "C" void kernel_launch(void* const* d_in, const int* in_sizes, int n_in,
                              void* d_out, int out_size, void* d_ws, size_t ws_size,
                              hipStream_t stream) {
    const float* prob  = (const float*)d_in[0];
    const float* gal   = (const float*)d_in[1];
    const float* bn_g  = (const float*)d_in[2];
    const float* bn_b  = (const float*)d_in[3];
    const float* fc_w  = (const float*)d_in[4];
    const float* fc_b  = (const float*)d_in[5];
    const float* lbn_g = (const float*)d_in[6];
    const float* lbn_b = (const float*)d_in[7];
    char*  ws   = (char*)d_ws;
    float* part = (float*)(ws + OFF_PART);

    repack_kernel<<<384, 256, 0, stream>>>(prob, gal, ws);           // 98304 frag-lanes
    score_kernel<<<NPAIR, 256, 0, stream>>>(fc_w, ws, part);         // 2304 pair blocks
    finish_kernel<<<1, 256, 0, stream>>>(part, fc_w, bn_g, bn_b, fc_b,
                                         lbn_g, lbn_b, (float*)d_out);
}
